// TreeLSTM_64742337020080
// MI455X (gfx1250) — matrix operand order repaired and checked by reading, not compiled, not run on hardware
//
#include <hip/hip_runtime.h>
#include <hip/hip_bf16.h>

// ---------------------------------------------------------------------------
// CDNA5 (gfx1250) TreeLSTM implementation: all GEMMs via v_wmma_f32_16x16x32_f16
// ---------------------------------------------------------------------------

typedef _Float16 half_t;
typedef __attribute__((ext_vector_type(16))) _Float16 v16h;
typedef __attribute__((ext_vector_type(8)))  _Float16 v8h;
typedef __attribute__((ext_vector_type(8)))  float    v8f;

#define NN 9841
#define HDIM 512
#define EDIM 256
#define SMAX 2187
#define SPMAX 2208
#define TMAX 6561
#define OUTW 12648

__device__ __forceinline__ int imin(int a, int b) { return a < b ? a : b; }
__device__ __forceinline__ float sigf(float x) { return 1.0f / (1.0f + __expf(-x)); }

// --------------------------- generic WMMA GEMM -----------------------------
// C[M,N] = act( A[M,K] (f16 rows) * W[N,K]^T (f16 rows) + bias[N] + D[M,N] )
// batched over gridDim.z with element strides sA/sW/sD/sC/sCh.
// Block = 128 threads = 4 waves (2x2); each wave computes a 32x64 tile
// (2x4 f32 accumulators) -> block tile 64x128. 12 b128 loads : 8 wmma per K-step.
struct GemmArgs {
    const half_t* A; const half_t* W; const float* bias; const float* D;
    float* C; half_t* Ch;
    long sA, sW, sD, sC, sCh;
    int lda, ldw, ldd, ldc, ldch;
    int M, N, K, act, nt;
};

__device__ __forceinline__ v16h frag_a(const half_t* p, int aoff) {
    // A 16x32 f16 layout: lanes<16 hold K {0..7, 16..23}; lanes>=16 hold K {8..15, 24..31}
    v8h x0 = *(const v8h*)(p + aoff);
    v8h x1 = *(const v8h*)(p + aoff + 16);
    v16h f;
#pragma unroll
    for (int j = 0; j < 8; ++j) { f[j] = x0[j]; f[j + 8] = x1[j]; }
    return f;
}
__device__ __forceinline__ v16h frag_b(const half_t* p) {
    // B 32x16 f16 layout: lane n<16 holds col n K {0..7, 16..23}; lane n>=16 holds col n-16 K {8..15, 24..31}
    v8h x0 = *(const v8h*)p;
    v8h x1 = *(const v8h*)(p + 16);
    v16h f;
#pragma unroll
    for (int j = 0; j < 8; ++j) { f[j] = x0[j]; f[j + 8] = x1[j]; }
    return f;
}

__global__ __launch_bounds__(128) void k_gemm(GemmArgs g) {
    const int lane = threadIdx.x & 31;
    const int wave = threadIdx.x >> 5;
    const int row0 = blockIdx.y * 64 + (wave >> 1) * 32;
    const int col0 = blockIdx.x * 128 + (wave & 1) * 64;
    if (row0 >= g.M || col0 >= g.N) return;  // wave-uniform
    const size_t z = blockIdx.z;
    const half_t* A = g.A + z * (size_t)g.sA;
    const half_t* W = g.W + z * (size_t)g.sW;
    const int m15 = lane & 15;
    const bool lo = lane < 16;
    const int aoff = lo ? 0 : 8;
    const int boff = lo ? 0 : 8;
    long ar[2], wr[4];
#pragma unroll
    for (int i = 0; i < 2; ++i) ar[i] = (long)imin(row0 + 16 * i + m15, g.M - 1) * g.lda;
#pragma unroll
    for (int j = 0; j < 4; ++j) wr[j] = (long)imin(col0 + 16 * j + m15, g.N - 1) * g.ldw;

    v8f acc[2][4];
#pragma unroll
    for (int i = 0; i < 2; ++i)
#pragma unroll
        for (int j = 0; j < 4; ++j) acc[i][j] = (v8f){};

    for (int k0 = 0; k0 < g.K; k0 += 32) {
        v16h a[2], b[4];
#pragma unroll
        for (int i = 0; i < 2; ++i) a[i] = frag_a(A + ar[i] + k0, aoff);
#pragma unroll
        for (int j = 0; j < 4; ++j) b[j] = frag_b(W + wr[j] + k0 + boff);
        if (k0 + 32 < g.K) {
            __builtin_prefetch(A + ar[0] + k0 + 32, 0, 1);
            __builtin_prefetch(W + wr[0] + k0 + 32, 0, 1);
        }
#pragma unroll
        for (int i = 0; i < 2; ++i)
#pragma unroll
            for (int j = 0; j < 4; ++j)
                acc[i][j] = __builtin_amdgcn_wmma_f32_16x16x32_f16(
                    false, a[i], false, b[j], (short)0, acc[i][j], false, false);
    }

    float* C = g.C + z * (size_t)g.sC;
    const float* D = g.D ? g.D + z * (size_t)g.sD : nullptr;
    half_t* Ch = g.Ch ? g.Ch + z * (size_t)g.sCh : nullptr;
#pragma unroll
    for (int i = 0; i < 2; ++i) {
#pragma unroll
        for (int j = 0; j < 4; ++j) {
            const int col = col0 + j * 16 + m15;
            if (col >= g.N) continue;
            const int rbase = row0 + i * 16 + (lo ? 0 : 8);
            const float bv = g.bias ? g.bias[col] : 0.0f;
#pragma unroll
            for (int v = 0; v < 8; ++v) {
                const int r = rbase + v;  // C layout: VGPR v -> M = v (+8 for hi lanes)
                if (r >= g.M) break;
                float val = acc[i][j][v] + bv;
                if (D) val += D[(size_t)r * g.ldd + col];
                if (g.act) val = fmaxf(val, 0.0f);
                if (g.nt) __builtin_nontemporal_store(val, &C[(size_t)r * g.ldc + col]);
                else C[(size_t)r * g.ldc + col] = val;
                if (Ch) Ch[(size_t)r * g.ldch + col] = (half_t)val;
            }
        }
    }
}

// ------------------------------ elementwise --------------------------------
__global__ void k_f32_to_f16(const float* a, half_t* b, long n) {
    long i = (long)blockIdx.x * 256 + threadIdx.x;
    if (i < n) b[i] = (half_t)a[i];
}

__global__ void k_embed(const float* Eu, const float* Ef, const float* Et,
                        const int* user, const int* feat, const int* tim,
                        const int* mask, const int* maskT, float* x, half_t* xh) {
    long i = (long)blockIdx.x * 256 + threadIdx.x;
    if (i >= (long)NN * EDIM) return;
    int nd = (int)(i >> 8), d = (int)(i & 255);
    int m = mask[nd];
    float v = (d < 128) ? Eu[(long)(user[nd] * m) * 128 + d]
                        : Ef[(long)(feat[nd] * m) * 128 + (d - 128)];
    v += 0.5f * Et[(long)(tim[nd] * maskT[nd]) * EDIM + d];
    v *= (float)m;
    x[i] = v; xh[i] = (half_t)v;
}

__global__ void k_leaf(const float* iouX, float* c, float* h, half_t* hh, int s0, int n) {
    long i = (long)blockIdx.x * 256 + threadIdx.x;
    if (i >= (long)n * HDIM) return;
    int s = (int)(i >> 9), d = (int)(i & 511);
    long g = (long)s0 + s;
    float iv = iouX[g * 1536 + d];
    float ov = iouX[g * 1536 + 512 + d];
    float uv = iouX[g * 1536 + 1024 + d];
    float cl = sigf(iv) * tanhf(uv);
    float hv = sigf(ov) * tanhf(cl);
    c[g * HDIM + d] = cl; h[g * HDIM + d] = hv; hh[g * HDIM + d] = (half_t)hv;
}

__global__ void k_gather_hc(const float* h, float* tok, half_t* tok16, int s0, int n) {
    long i = (long)blockIdx.x * 256 + threadIdx.x;
    if (i >= (long)n * 1536) return;
    int t = (int)(i >> 9), d = (int)(i & 511);
    int s = t / 3, j = t - 3 * s;
    long child = 3L * (s0 + s) + 1 + j;
    float v = h[child * HDIM + d];
    tok[i] = v; tok16[i] = (half_t)v;
}

__global__ void k_split_qk(const float* qkv, half_t* q, half_t* k, int S) {
    long i = (long)blockIdx.x * 256 + threadIdx.x;
    if (i >= 6L * S * 256) return;
    int z = (int)(i / ((long)S * 256));
    long rem = i - (long)z * S * 256;
    int t = (int)(rem >> 8), d = (int)(rem & 255);
    int b = z >> 1, hh = z & 1;
    long src = (long)(t * 3 + b) * 1536 + hh * 256 + d;
    q[i] = (half_t)qkv[src];
    k[i] = (half_t)qkv[src + 512];
}

__global__ void k_build_vT(const float* qkv, half_t* vT, int S, int SP) {
    long i = (long)blockIdx.x * 256 + threadIdx.x;
    if (i >= 6L * 256 * SP) return;
    int z = (int)(i / ((long)256 * SP));
    long rem = i - (long)z * 256 * SP;
    int d = (int)(rem / SP), t = (int)(rem % SP);
    int b = z >> 1, hh = z & 1;
    half_t v = (half_t)0.0f;
    if (t < S) v = (half_t)qkv[(long)(t * 3 + b) * 1536 + 1024 + hh * 256 + d];
    vT[i] = v;
}

// row-wise softmax over f32 scores, writes f16 P *in place* (half row stride 2*SP)
__global__ __launch_bounds__(256) void k_softmax(float* sc, int S, int SP, float scale) {
    __shared__ float buf[SPMAX];
    __shared__ float red[8];
    const int row = blockIdx.x;
    const size_t z = blockIdx.z;
    float* rowf = sc + z * (size_t)S * SP + (size_t)row * SP;
    const int tid = threadIdx.x;
    float lmax = -1e30f;
    for (int i = tid; i < S; i += 256) { float v = rowf[i] * scale; buf[i] = v; lmax = fmaxf(lmax, v); }
    for (int o = 16; o; o >>= 1) lmax = fmaxf(lmax, __shfl_xor(lmax, o, 32));
    if ((tid & 31) == 0) red[tid >> 5] = lmax;
    __syncthreads();
    float m = red[0];
#pragma unroll
    for (int i = 1; i < 8; ++i) m = fmaxf(m, red[i]);
    __syncthreads();
    float lsum = 0.0f;
    for (int i = tid; i < S; i += 256) { float e = __expf(buf[i] - m); buf[i] = e; lsum += e; }
    for (int o = 16; o; o >>= 1) lsum += __shfl_xor(lsum, o, 32);
    if ((tid & 31) == 0) red[tid >> 5] = lsum;
    __syncthreads();
    float s = 0.0f;
#pragma unroll
    for (int i = 0; i < 8; ++i) s += red[i];
    const float inv = 1.0f / s;
    half_t* rowh = (half_t*)rowf;
    for (int i = tid; i < SP; i += 256) rowh[i] = (i < S) ? (half_t)(buf[i] * inv) : (half_t)0.0f;
}

// y = LN(x + r) * gamma + beta ; one wave per row (wave32, 16 elems/lane)
__global__ __launch_bounds__(256) void k_add_ln(const float* x, const float* r,
                                                const float* gam, const float* bet,
                                                float* y, half_t* yh, int T) {
    const int row = blockIdx.x * 8 + (threadIdx.x >> 5);
    if (row >= T) return;
    const int lane = threadIdx.x & 31;
    const float* px = x + (size_t)row * HDIM;
    const float* pr = r + (size_t)row * HDIM;
    float v[16]; float s = 0.0f;
#pragma unroll
    for (int i = 0; i < 16; ++i) { int d = lane + i * 32; v[i] = px[d] + pr[d]; s += v[i]; }
    for (int o = 16; o; o >>= 1) s += __shfl_xor(s, o, 32);
    const float mean = s * (1.0f / 512.0f);
    float vs = 0.0f;
#pragma unroll
    for (int i = 0; i < 16; ++i) { float dl = v[i] - mean; vs += dl * dl; }
    for (int o = 16; o; o >>= 1) vs += __shfl_xor(vs, o, 32);
    const float rstd = rsqrtf(vs * (1.0f / 512.0f) + 1e-5f);
#pragma unroll
    for (int i = 0; i < 16; ++i) {
        int d = lane + i * 32;
        float o = (v[i] - mean) * rstd * gam[d] + bet[d];
        y[(size_t)row * HDIM + d] = o;
        yh[(size_t)row * HDIM + d] = (half_t)o;
    }
}

__global__ void k_update(const float* iou, const float* hf, const float* wfX, const float* bf,
                         float* c, float* h, half_t* hh, int s0, int n) {
    long i = (long)blockIdx.x * 256 + threadIdx.x;
    if (i >= (long)n * HDIM) return;
    int s = (int)(i >> 9), d = (int)(i & 511);
    long g = (long)s0 + s;
    float wf = wfX[g * HDIM + d] + bf[d];
    float csum = 0.0f;
#pragma unroll
    for (int j = 0; j < 3; ++j) {
        float f = sigf(wf + hf[(long)s * 1536 + j * 512 + d]);
        csum += f * c[(3 * g + 1 + j) * HDIM + d];
    }
    float iv = iou[(long)s * 1536 + d];
    float ov = iou[(long)s * 1536 + 512 + d];
    float uv = iou[(long)s * 1536 + 1024 + d];
    float cn = sigf(iv) * tanhf(uv) + csum;
    float hv = sigf(ov) * tanhf(cn);
    c[g * HDIM + d] = cn; h[g * HDIM + d] = hv; hh[g * HDIM + d] = (half_t)hv;
}

// ------------------------------- host side ---------------------------------
struct EncP { const float *W1,*W2,*Wo,*Wqkv,*b1,*b2,*bn1,*bn2,*bo,*bqkv,*g1,*g2; };
struct CellP { const float *Uiou,*Wf,*Wiou,*bf,*biou; EncP enc[2]; };
struct DecP { const float *Wcat,*Wcoo,*Wpoi,*bcat,*bcoo,*bpoi; };
struct EncH { half_t *Wqkv,*Wo,*W1,*W2; };
struct CellH { half_t *Uiou,*Wf,*Wiou; EncH enc[2]; };
struct DecH { half_t *Wpoi,*Wcat,*Wcoo; };

extern "C" void kernel_launch(void* const* d_in, const int* in_sizes, int n_in,
                              void* d_out, int out_size, void* d_ws, size_t ws_size,
                              hipStream_t stream) {
    (void)in_sizes; (void)n_in; (void)out_size; (void)ws_size;
    auto F = [&](int i) { return (const float*)d_in[i]; };
    auto I = [&](int i) { return (const int*)d_in[i]; };

    // pytree (sorted-key) flatten order of setup_inputs dict
    const int* features_in  = I(0);
    const int* features_out = I(1);
    const int* mask_in      = I(2);
    const int* mask_out     = I(3);
    const float* E_fuse   = F(4);
    const float* E_fuse_o = F(5);
    const float* E_time   = F(6);
    const float* E_time_o = F(7);
    const float* E_user   = F(8);
    const float* E_user_o = F(9);
    auto encAt = [&](int b) { EncP e; e.W1=F(b); e.W2=F(b+1); e.Wo=F(b+2); e.Wqkv=F(b+3);
        e.b1=F(b+4); e.b2=F(b+5); e.bn1=F(b+6); e.bn2=F(b+7); e.bo=F(b+8); e.bqkv=F(b+9);
        e.g1=F(b+10); e.g2=F(b+11); return e; };
    auto cellAt = [&](int b) { CellP c; c.Uiou=F(b); c.Wf=F(b+1); c.Wiou=F(b+2);
        c.bf=F(b+3); c.biou=F(b+4); c.enc[0]=encAt(b+5); c.enc[1]=encAt(b+17); return c; };
    auto decAt = [&](int b) { DecP d; d.Wcat=F(b); d.Wcoo=F(b+1); d.Wpoi=F(b+2);
        d.bcat=F(b+3); d.bcoo=F(b+4); d.bpoi=F(b+5); return d; };
    CellP cell   = cellAt(10);
    CellP cell_o = cellAt(39);
    DecP dec     = decAt(68);
    DecP dec_o   = decAt(74);
    const int* time_in  = I(80);
    const int* time_out = I(81);
    const int* user_in  = I(82);
    const int* user_out = I(83);

    // ---- workspace bump allocator ----
    size_t off = 0;
    auto alloc = [&](size_t bytes) { void* r = (char*)d_ws + off; off += (bytes + 255) & ~(size_t)255; return r; };
    auto af = [&](size_t n) { return (float*)alloc(n * 4); };
    auto ah = [&](size_t n) { return (half_t*)alloc(n * 2); };
    auto conv = [&](const float* src, size_t n) {
        half_t* dst = ah(n);
        k_f32_to_f16<<<(int)((n + 255) / 256), 256, 0, stream>>>(src, dst, (long)n);
        return dst;
    };

    // f16 weights (fit in 192MB L2 -> B operand is L2 resident)
    auto cellConv = [&](const CellP& c) {
        CellH h;
        h.Uiou = conv(c.Uiou, 1536 * 1536);
        h.Wf   = conv(c.Wf,   512 * 256);
        h.Wiou = conv(c.Wiou, 1536 * 256);
        for (int l = 0; l < 2; ++l) {
            h.enc[l].Wqkv = conv(c.enc[l].Wqkv, 1536 * 512);
            h.enc[l].Wo   = conv(c.enc[l].Wo,   512 * 512);
            h.enc[l].W1   = conv(c.enc[l].W1,   512 * 512);
            h.enc[l].W2   = conv(c.enc[l].W2,   512 * 512);
        }
        return h;
    };
    auto decConv = [&](const DecP& d) {
        DecH h;
        h.Wpoi = conv(d.Wpoi, 5000 * 512);
        h.Wcat = conv(d.Wcat,  300 * 512);
        h.Wcoo = conv(d.Wcoo, 1024 * 512);
        return h;
    };
    CellH cellH = cellConv(cell), cellH_o = cellConv(cell_o);
    DecH decH = decConv(dec), decH_o = decConv(dec_o);

    // activations
    float* xin   = af((size_t)NN * EDIM);   half_t* xin16  = ah((size_t)NN * EDIM);
    float* xout  = af((size_t)NN * EDIM);   half_t* xout16 = ah((size_t)NN * EDIM);
    float* iouXi = af((size_t)NN * 1536);   float* iouXo   = af((size_t)NN * 1536);
    float* wfXi  = af((size_t)NN * HDIM);   float* wfXo    = af((size_t)NN * HDIM);
    float* hin   = af((size_t)NN * HDIM);   float* cin     = af((size_t)NN * HDIM);
    float* hout  = af((size_t)NN * HDIM);   float* cout_   = af((size_t)NN * HDIM);
    half_t* hin16  = ah((size_t)NN * HDIM);
    half_t* hout16 = ah((size_t)NN * HDIM);
    float* tok   = af((size_t)TMAX * HDIM); half_t* tok16  = ah((size_t)TMAX * HDIM);
    float* qkv   = af((size_t)TMAX * 1536);
    half_t* qh   = ah(6L * SMAX * 256);
    half_t* kh   = ah(6L * SMAX * 256);
    half_t* vth  = ah(6L * 256 * SPMAX);
    float* scores = af(6L * SMAX * SPMAX);
    float* attn  = af((size_t)TMAX * HDIM); half_t* attn16 = ah((size_t)TMAX * HDIM);
    float* aout  = af((size_t)TMAX * HDIM);
    float* ffa   = af((size_t)TMAX * HDIM); half_t* ffa16  = ah((size_t)TMAX * HDIM);
    float* ffb   = af((size_t)TMAX * HDIM);
    float* ioub  = af((size_t)SMAX * 1536);
    float* dout  = (float*)d_out;

    auto gemm = [&](const half_t* A, long sA, int lda,
                    const half_t* W, long sW, int ldw,
                    const float* bias, const float* D, long sD, int ldd,
                    float* C, long sC, int ldc, half_t* Ch, long sCh, int ldch,
                    int M, int N, int K, int act, int batch, int nt) {
        GemmArgs ga{A, W, bias, D, C, Ch, sA, sW, sD, sC, sCh,
                    lda, ldw, ldd, ldc, ldch, M, N, K, act, nt};
        dim3 gr((N + 127) / 128, (M + 63) / 64, batch);
        k_gemm<<<gr, 128, 0, stream>>>(ga);
    };

    // ---- embeddings (note out-tree replicates the mask_in time-index quirk) ----
    int embBlocks = (NN * EDIM + 255) / 256;
    k_embed<<<embBlocks, 256, 0, stream>>>(E_user, E_fuse, E_time, user_in, features_in,
                                           time_in, mask_in, mask_in, xin, xin16);
    k_embed<<<embBlocks, 256, 0, stream>>>(E_user_o, E_fuse_o, E_time_o, user_out, features_out,
                                           time_out, mask_out, mask_in, xout, xout16);

    static const int starts[10] = {0, 1, 4, 13, 40, 121, 364, 1093, 3280, 9841};

    auto run_tree = [&](const CellP& cp, const CellH& chh, half_t* x16,
                        float* iouX, float* wfX, float* hb, float* cb, half_t* hb16) {
        gemm(x16, 0, EDIM, chh.Wiou, 0, EDIM, nullptr, nullptr, 0, 0,
             iouX, 0, 1536, nullptr, 0, 0, NN, 1536, EDIM, 0, 1, 0);
        gemm(x16, 0, EDIM, chh.Wf, 0, EDIM, nullptr, nullptr, 0, 0,
             wfX, 0, HDIM, nullptr, 0, 0, NN, HDIM, EDIM, 0, 1, 0);
        // leaves
        k_leaf<<<(6561 * HDIM + 255) / 256, 256, 0, stream>>>(iouX, cb, hb, hb16, 3280, 6561);
        for (int l = 7; l >= 0; --l) {
            const int s0 = starts[l];
            const int n = starts[l + 1] - starts[l];
            const int S = n, T = 3 * n, SP = (S + 31) & ~31;
            k_gather_hc<<<((long)n * 1536 + 255) / 256, 256, 0, stream>>>(hb, tok, tok16, s0, n);
            for (int li = 0; li < 2; ++li) {
                const EncP& ep = cp.enc[li];
                const EncH& eh = chh.enc[li];
                gemm(tok16, 0, HDIM, eh.Wqkv, 0, HDIM, ep.bqkv, nullptr, 0, 0,
                     qkv, 0, 1536, nullptr, 0, 0, T, 1536, HDIM, 0, 1, 0);
                k_split_qk<<<(int)((6L * S * 256 + 255) / 256), 256, 0, stream>>>(qkv, qh, kh, S);
                k_build_vT<<<(int)((6L * 256 * SP + 255) / 256), 256, 0, stream>>>(qkv, vth, S, SP);
                // scores[b,h] = q @ k^T  (batched over 6)
                gemm(qh, (long)S * 256, 256, kh, (long)S * 256, 256, nullptr, nullptr, 0, 0,
                     scores, (long)S * SP, SP, nullptr, 0, 0, S, S, 256, 0, 6, 0);
                k_softmax<<<dim3(S, 1, 6), 256, 0, stream>>>(scores, S, SP, 0.0625f);
                // ctx = P @ V, written straight into token layout (offset z*256, ldc 1536)
                gemm((half_t*)scores, 2L * S * SP, 2 * SP, vth, 256L * SP, SP,
                     nullptr, nullptr, 0, 0, attn, 256, 1536, attn16, 256, 1536,
                     S, 256, SP, 0, 6, 0);
                gemm(attn16, 0, HDIM, eh.Wo, 0, HDIM, ep.bo, nullptr, 0, 0,
                     aout, 0, HDIM, nullptr, 0, 0, T, HDIM, HDIM, 0, 1, 0);
                k_add_ln<<<(T + 7) / 8, 256, 0, stream>>>(tok, aout, ep.g1, ep.bn1, tok, tok16, T);
                gemm(tok16, 0, HDIM, eh.W1, 0, HDIM, ep.b1, nullptr, 0, 0,
                     ffa, 0, HDIM, ffa16, 0, HDIM, T, HDIM, HDIM, 1, 1, 0);
                gemm(ffa16, 0, HDIM, eh.W2, 0, HDIM, ep.b2, nullptr, 0, 0,
                     ffb, 0, HDIM, nullptr, 0, 0, T, HDIM, HDIM, 0, 1, 0);
                k_add_ln<<<(T + 7) / 8, 256, 0, stream>>>(tok, ffb, ep.g2, ep.bn2, tok, tok16, T);
            }
            // iou = x@Wiou^T + hf@Uiou^T + b_iou   (hf = token buffer viewed [n,1536])
            gemm(tok16, 0, 1536, chh.Uiou, 0, 1536, cp.biou,
                 iouX + (size_t)s0 * 1536, 0, 1536,
                 ioub, 0, 1536, nullptr, 0, 0, n, 1536, 1536, 0, 1, 0);
            k_update<<<((long)n * HDIM + 255) / 256, 256, 0, stream>>>(
                ioub, tok, wfX, cp.bf, cb, hb, hb16, s0, n);
        }
    };

    run_tree(cell,   cellH,   xin16,  iouXi, wfXi, hin,  cin,   hin16);
    run_tree(cell_o, cellH_o, xout16, iouXo, wfXo, hout, cout_, hout16);

    // ---- decoders straight into d_out (ldc = 12648), non-temporal stores:
    // 498MB written once, never re-read -> stream past L2, keep f16 weights hot.
    auto decode = [&](const DecP& dp, const DecH& dh, half_t* hh, int colBase) {
        gemm(hh, 0, HDIM, dh.Wpoi, 0, HDIM, dp.bpoi, nullptr, 0, 0,
             dout + colBase, 0, OUTW, nullptr, 0, 0, NN, 5000, HDIM, 0, 1, 1);
        gemm(hh, 0, HDIM, dh.Wcat, 0, HDIM, dp.bcat, nullptr, 0, 0,
             dout + colBase + 5000, 0, OUTW, nullptr, 0, 0, NN, 300, HDIM, 0, 1, 1);
        gemm(hh, 0, HDIM, dh.Wcoo, 0, HDIM, dp.bcoo, nullptr, 0, 0,
             dout + colBase + 5300, 0, OUTW, nullptr, 0, 0, NN, 1024, HDIM, 0, 1, 1);
    };
    decode(dec,   decH,   hin16,  0);
    decode(dec_o, decH_o, hout16, 6324);
}
